// MambaEncoderFFN_91199335563755
// MI455X (gfx1250) — matrix operand order repaired and checked by reading, not compiled, not run on hardware
//
#include <hip/hip_runtime.h>
#include <hip/hip_bf16.h>

// ---------------- problem constants ----------------
static constexpr int BQ      = 4;
static constexpr int LQ      = 1024;
static constexpr int DM      = 512;
static constexpr int NLAYERS = 6;
static constexpr int HEADDIM = 64;
static constexpr int NSTATE  = 16;
static constexpr int DCONV   = 4;
static constexpr int DI      = 2 * DM;              // 1024
static constexpr int NH      = DI / HEADDIM;        // 16
static constexpr int CONV_DIM= DI + 2 * NSTATE;     // 1056
static constexpr int DPROJ   = 2 * DI + 2 * NSTATE + NH; // 2096
static constexpr int DFF     = 2048;
static constexpr int MTOK    = BQ * LQ;             // 4096 rows
static constexpr float EPS   = 1e-5f;

typedef __attribute__((ext_vector_type(16))) __bf16 v16bf;
typedef __attribute__((ext_vector_type(8)))  __bf16 v8bf;
typedef __attribute__((ext_vector_type(8)))  float  v8f;

// ---------------- f32 -> bf16 conversion (weights) ----------------
__global__ void cvt_f32_bf16(const float* __restrict__ in, __bf16* __restrict__ out, size_t n) {
    size_t i = (size_t)blockIdx.x * blockDim.x + threadIdx.x;
    size_t stride = (size_t)gridDim.x * blockDim.x;
    for (; i < n; i += stride) out[i] = (__bf16)in[i];
}

// ---------------- RMSNorm (f32 in, bf16 out) ----------------
__global__ __launch_bounds__(256) void rmsnorm_bf16(const float* __restrict__ x,
                                                    const float* __restrict__ w,
                                                    __bf16* __restrict__ out, int dim) {
    int row = blockIdx.x;
    const float* xr = x + (size_t)row * dim;
    float s = 0.f;
    for (int i = threadIdx.x; i < dim; i += 256) { float v = xr[i]; s += v * v; }
    __shared__ float sm[256];
    sm[threadIdx.x] = s; __syncthreads();
    for (int o = 128; o > 0; o >>= 1) {
        if (threadIdx.x < o) sm[threadIdx.x] += sm[threadIdx.x + o];
        __syncthreads();
    }
    float rs = rsqrtf(sm[0] / dim + EPS);
    __bf16* orow = out + (size_t)row * dim;
    for (int i = threadIdx.x; i < dim; i += 256) orow[i] = (__bf16)(xr[i] * rs * w[i]);
}

// ---------------- final RMSNorm (f32 out) ----------------
__global__ __launch_bounds__(256) void rmsnorm_f32(const float* __restrict__ x,
                                                   const float* __restrict__ w,
                                                   float* __restrict__ out, int dim) {
    int row = blockIdx.x;
    const float* xr = x + (size_t)row * dim;
    float s = 0.f;
    for (int i = threadIdx.x; i < dim; i += 256) { float v = xr[i]; s += v * v; }
    __shared__ float sm[256];
    sm[threadIdx.x] = s; __syncthreads();
    for (int o = 128; o > 0; o >>= 1) {
        if (threadIdx.x < o) sm[threadIdx.x] += sm[threadIdx.x + o];
        __syncthreads();
    }
    float rs = rsqrtf(sm[0] / dim + EPS);
    float* orow = out + (size_t)row * dim;
    for (int i = threadIdx.x; i < dim; i += 256) orow[i] = xr[i] * rs * w[i];
}

// ---------------- gate + group RMSNorm: out = rmsnorm(y * silu(z), gw) -> bf16 ----------------
__global__ __launch_bounds__(256) void gate_gnorm_bf16(const float* __restrict__ y,
                                                       const float* __restrict__ zx, // (M, DPROJ), z = cols [0,DI)
                                                       const float* __restrict__ gw,
                                                       __bf16* __restrict__ out) {
    int row = blockIdx.x;
    const float* yr = y + (size_t)row * DI;
    const float* zr = zx + (size_t)row * DPROJ;
    float vloc[4];
    float s = 0.f;
    #pragma unroll
    for (int j = 0; j < 4; ++j) {
        int i = threadIdx.x + j * 256;
        float z = zr[i];
        float v = yr[i] * (z / (1.f + __expf(-z)));
        vloc[j] = v;
        s += v * v;
    }
    __shared__ float sm[256];
    sm[threadIdx.x] = s; __syncthreads();
    for (int o = 128; o > 0; o >>= 1) {
        if (threadIdx.x < o) sm[threadIdx.x] += sm[threadIdx.x + o];
        __syncthreads();
    }
    float rs = rsqrtf(sm[0] / (float)DI + EPS);
    __bf16* orow = out + (size_t)row * DI;
    #pragma unroll
    for (int j = 0; j < 4; ++j) {
        int i = threadIdx.x + j * 256;
        orow[i] = (__bf16)(vloc[j] * rs * gw[i]);
    }
}

// ---------------- depthwise causal conv (DCONV=4) + bias + SiLU ----------------
__global__ void conv_silu(const float* __restrict__ zx,  // (M, DPROJ), xBC = cols [DI, DI+CONV_DIM)
                          const float* __restrict__ cw,  // (CONV_DIM, DCONV)
                          const float* __restrict__ cb,  // (CONV_DIM)
                          float* __restrict__ out) {     // (M, CONV_DIM)
    size_t idx = (size_t)blockIdx.x * blockDim.x + threadIdx.x;
    if (idx >= (size_t)MTOK * CONV_DIM) return;
    int c = (int)(idx % CONV_DIM);
    int m = (int)(idx / CONV_DIM);
    int l = m % LQ;
    float acc = cb[c];
    #pragma unroll
    for (int t = 0; t < DCONV; ++t) {
        int ls = l - (DCONV - 1) + t;
        if (ls >= 0)
            acc += zx[(size_t)(m - (DCONV - 1 - t)) * DPROJ + DI + c] * cw[c * DCONV + t];
    }
    out[(size_t)m * CONV_DIM + c] = acc / (1.f + __expf(-acc));  // silu
}

// ---------------- dt = softplus(raw + dtb); dA = exp(-exp(Alog)*dt) ----------------
__global__ void dt_da(const float* __restrict__ zx, const float* __restrict__ dtb,
                      const float* __restrict__ Alog,
                      float* __restrict__ dt, float* __restrict__ dA) {
    int idx = blockIdx.x * blockDim.x + threadIdx.x;
    if (idx >= MTOK * NH) return;
    int h = idx % NH, m = idx / NH;
    float raw = zx[(size_t)m * DPROJ + (DPROJ - NH) + h] + dtb[h];
    float d = raw > 20.f ? raw : log1pf(__expf(raw));
    dt[idx] = d;
    dA[idx] = __expf(-__expf(Alog[h]) * d);
}

// ---------------- sequential SSM scan: one block per (b, head) ----------------
__global__ __launch_bounds__(64) void ssm_scan(const float* __restrict__ xbc, // (M, CONV_DIM)
                                               const float* __restrict__ dt,  // (M, NH)
                                               const float* __restrict__ dA,  // (M, NH)
                                               const float* __restrict__ Dh,  // (NH)
                                               float* __restrict__ y) {       // (M, DI)
    int bh = blockIdx.x;
    int b = bh >> 4, h = bh & 15;
    int p = threadIdx.x;                 // 0..63 head-dim element
    float hs[NSTATE];
    #pragma unroll
    for (int n = 0; n < NSTATE; ++n) hs[n] = 0.f;
    __shared__ float sB[NSTATE], sC[NSTATE], sScal[2];
    float Dv = Dh[h];
    for (int l = 0; l < LQ; ++l) {
        size_t m = (size_t)b * LQ + l;
        const float* row = xbc + m * CONV_DIM;
        if (p < 16)       sB[p]      = row[DI + p];
        else if (p < 32)  sC[p - 16] = row[DI + p];
        else if (p == 32) sScal[0]   = dt[m * NH + h];
        else if (p == 33) sScal[1]   = dA[m * NH + h];
        __syncthreads();
        float xv = row[(size_t)h * HEADDIM + p];
        float a = sScal[1];
        float coef = sScal[0] * xv;
        float acc = Dv * xv;
        #pragma unroll
        for (int n = 0; n < NSTATE; ++n) {
            hs[n] = hs[n] * a + coef * sB[n];
            acc += hs[n] * sC[n];
        }
        y[m * DI + (size_t)h * HEADDIM + p] = acc;
        __syncthreads();
    }
}

// ---------------- WMMA GEMM: C(M,N) = A(M,K) @ W(N,K)^T  [+bias][gelu][+res] ----------------
// Block tile 128x128, 8 waves: 4 along M (32 rows each), 2 along N (64 cols each).
// The shared 128x32 weight tile is staged into LDS with async global->LDS copies
// (double-buffered, overlapped with WMMA on the other buffer, tracked by ASYNCcnt).
// LDS row stride = 40 bf16 (80B) -> 20-dword lane stride, gcd(20,64)=4: conflict-free
// across each 16-lane half reading B fragments.
template<bool GELU, bool BIAS, bool RES, bool OUTBF>
__global__ __launch_bounds__(256) void gemm_wmma_bf16(
    const __bf16* __restrict__ A, const __bf16* __restrict__ Wt,
    const float* __restrict__ bias, const float* __restrict__ res,
    float* __restrict__ Cf, __bf16* __restrict__ Cb,
    int Nr, int Kr) {
    constexpr int BROW = 40;                 // padded LDS row stride (elements)
    __shared__ __bf16 sBt[2][128 * BROW];    // 2 x 10 KB

    const int lane = threadIdx.x & 31;
    const int wave = threadIdx.x >> 5;
    const int wm   = wave & 3;          // 0..3 (M)
    const int wn   = wave >> 2;         // 0..1 (N)
    const int lr   = lane & 15;
    const int lh   = lane >> 4;
    const int Mblk = blockIdx.y * 128;
    const int Nblk = blockIdx.x * 128;

    v8f acc[2][4] = {};

    const int arow = Mblk + wm * 32 + lr;
    const __bf16* Abase = A + (size_t)arow * Kr;
    const int KT = Kr >> 5;             // K-tiles of 32

    // async-stage the 128x32 B tile for k-tile kt into LDS buffer `buf`
    auto stage_B = [&](int kt, int buf) {
        const int k0 = kt << 5;
        #pragma unroll
        for (int j = 0; j < 2; ++j) {
            int c   = (int)threadIdx.x + j * 256;   // 512 x 16B chunks
            int row = c >> 2;
            int sub = c & 3;
            int n = Nblk + row; if (n >= Nr) n = Nr - 1;   // clamp (junk rows never read for stores)
            const __bf16* g = Wt + (size_t)n * Kr + k0 + sub * 8;
            unsigned lo = (unsigned)(size_t)&sBt[buf][row * BROW + sub * 8];
            asm volatile("global_load_async_to_lds_b128 %0, %1, off"
                         :: "v"(lo), "v"(g) : "memory");
        }
    };

    stage_B(0, 0);
    for (int kt = 0; kt < KT; ++kt) {
        const int buf = kt & 1;
        asm volatile("s_wait_asynccnt 0x0" ::: "memory");  // my copies into `buf` landed
        __syncthreads();                                   // all waves' copies landed; prev reads of buf^1 done
        if (kt + 1 < KT) stage_B(kt + 1, buf ^ 1);         // overlap next tile copy with compute

        const int k0 = kt << 5;
        union { v16bf v; v8bf h[2]; } a[2], b[4];
        // A 16x32 bf16 fragment: lane holds K octets [lh*8, lh*8+8) and [16+lh*8, ...)
        #pragma unroll
        for (int mi = 0; mi < 2; ++mi) {
            const __bf16* ap = Abase + (size_t)(mi * 16) * Kr + k0;
            a[mi].h[0] = *(const v8bf*)(ap + lh * 8);
            a[mi].h[1] = *(const v8bf*)(ap + 16 + lh * 8);
        }
        // B 32x16 bf16 fragment from LDS: lane = column n, K octets [lh*8, lh*8+8) and [16+lh*8, ...)
        #pragma unroll
        for (int ni = 0; ni < 4; ++ni) {
            const __bf16* lp = &sBt[buf][(wn * 64 + ni * 16 + lr) * BROW + lh * 8];
            b[ni].h[0] = *(const v8bf*)lp;
            b[ni].h[1] = *(const v8bf*)(lp + 16);
        }
        #pragma unroll
        for (int mi = 0; mi < 2; ++mi)
            #pragma unroll
            for (int ni = 0; ni < 4; ++ni)
                acc[mi][ni] = __builtin_amdgcn_wmma_f32_16x16x32_bf16(
                    false, a[mi].v, false, b[ni].v, (short)0, acc[mi][ni], false, false);
    }

    // Epilogue. C tile layout: VGPR r -> M = r + 8*lh; lane -> N = lr.
    #pragma unroll
    for (int mi = 0; mi < 2; ++mi) {
        #pragma unroll
        for (int ni = 0; ni < 4; ++ni) {
            int n = Nblk + wn * 64 + ni * 16 + lr;
            if (n < Nr) {
                float bs = BIAS ? bias[n] : 0.f;
                #pragma unroll
                for (int r = 0; r < 8; ++r) {
                    int m = Mblk + wm * 32 + mi * 16 + r + lh * 8;
                    float v = acc[mi][ni][r] + bs;
                    if (GELU) v = 0.5f * v * (1.f + erff(v * 0.70710678118f));
                    size_t idx = (size_t)m * Nr + n;
                    if (RES) v += res[idx];
                    if (OUTBF) Cb[idx] = (__bf16)v;
                    else       Cf[idx] = v;
                }
            }
        }
    }
}

// ---------------- host driver ----------------
extern "C" void kernel_launch(void* const* d_in, const int* in_sizes, int n_in,
                              void* d_out, int out_size, void* d_ws, size_t ws_size,
                              hipStream_t stream) {
    const float* in_x      = (const float*)d_in[0];
    const float* in_proj_w = (const float*)d_in[1];
    const float* conv_w    = (const float*)d_in[2];
    const float* conv_b    = (const float*)d_in[3];
    const float* dt_bias   = (const float*)d_in[4];
    const float* A_log     = (const float*)d_in[5];
    const float* D_ssm     = (const float*)d_in[6];
    const float* gnorm_w   = (const float*)d_in[7];
    const float* out_proj_w= (const float*)d_in[8];
    const float* ffn_w1    = (const float*)d_in[9];
    const float* ffn_b1    = (const float*)d_in[10];
    const float* ffn_w2    = (const float*)d_in[11];
    const float* ffn_b2    = (const float*)d_in[12];
    const float* nm_w      = (const float*)d_in[13];
    const float* nf_w      = (const float*)d_in[14];
    const float* final_w   = (const float*)d_in[15];

    size_t off = 0;
    auto take = [&](size_t bytes) -> void* {
        void* p = (char*)d_ws + off;
        off = (off + bytes + 255) & ~(size_t)255;
        return p;
    };
    float*  X   = (float*) take((size_t)MTOK * DM * 4);
    __bf16* XN  = (__bf16*)take((size_t)MTOK * DM * 2);
    float*  ZX  = (float*) take((size_t)MTOK * DPROJ * 4);
    float*  XBC = (float*) take((size_t)MTOK * CONV_DIM * 4);
    float*  DT  = (float*) take((size_t)MTOK * NH * 4);
    float*  DA  = (float*) take((size_t)MTOK * NH * 4);
    float*  Y   = (float*) take((size_t)MTOK * DI * 4);
    __bf16* YB  = (__bf16*)take((size_t)MTOK * DI * 2);
    __bf16* HB  = (__bf16*)take((size_t)MTOK * DFF * 2);
    __bf16* WIN = (__bf16*)take((size_t)NLAYERS * DPROJ * DM * 2);
    __bf16* WOUT= (__bf16*)take((size_t)NLAYERS * DM * DI * 2);
    __bf16* W1B = (__bf16*)take((size_t)NLAYERS * DFF * DM * 2);
    __bf16* W2B = (__bf16*)take((size_t)NLAYERS * DM * DFF * 2);

    // live activation buffer = input x (residual stream), updated in place
    hipMemcpyAsync(X, in_x, (size_t)MTOK * DM * 4, hipMemcpyDeviceToDevice, stream);

    // weight conversions (deterministic, every call)
    cvt_f32_bf16<<<2048, 256, 0, stream>>>(in_proj_w,  WIN, (size_t)NLAYERS * DPROJ * DM);
    cvt_f32_bf16<<<2048, 256, 0, stream>>>(out_proj_w, WOUT,(size_t)NLAYERS * DM * DI);
    cvt_f32_bf16<<<2048, 256, 0, stream>>>(ffn_w1,     W1B, (size_t)NLAYERS * DFF * DM);
    cvt_f32_bf16<<<2048, 256, 0, stream>>>(ffn_w2,     W2B, (size_t)NLAYERS * DM * DFF);

    const dim3 blk(256);
    for (int i = 0; i < NLAYERS; ++i) {
        // ---- Mamba2 block ----
        rmsnorm_bf16<<<MTOK, blk, 0, stream>>>(X, nm_w + (size_t)i * DM, XN, DM);
        gemm_wmma_bf16<false,false,false,false>
            <<<dim3((DPROJ + 127) / 128, MTOK / 128), blk, 0, stream>>>(
                XN, WIN + (size_t)i * DPROJ * DM, nullptr, nullptr, ZX, nullptr, DPROJ, DM);
        dt_da<<<(MTOK * NH + 255) / 256, blk, 0, stream>>>(
            ZX, dt_bias + (size_t)i * NH, A_log + (size_t)i * NH, DT, DA);
        conv_silu<<<((size_t)MTOK * CONV_DIM + 255) / 256, blk, 0, stream>>>(
            ZX, conv_w + (size_t)i * CONV_DIM * DCONV, conv_b + (size_t)i * CONV_DIM, XBC);
        ssm_scan<<<BQ * NH, 64, 0, stream>>>(XBC, DT, DA, D_ssm + (size_t)i * NH, Y);
        gate_gnorm_bf16<<<MTOK, blk, 0, stream>>>(Y, ZX, gnorm_w + (size_t)i * DI, YB);
        gemm_wmma_bf16<false,false,true,false>
            <<<dim3(DM / 128, MTOK / 128), blk, 0, stream>>>(
                YB, WOUT + (size_t)i * DM * DI, nullptr, X, X, nullptr, DM, DI);
        // ---- FFN block ----
        rmsnorm_bf16<<<MTOK, blk, 0, stream>>>(X, nf_w + (size_t)i * DM, XN, DM);
        gemm_wmma_bf16<true,true,false,true>
            <<<dim3(DFF / 128, MTOK / 128), blk, 0, stream>>>(
                XN, W1B + (size_t)i * DFF * DM, ffn_b1 + (size_t)i * DFF, nullptr,
                nullptr, HB, DFF, DM);
        gemm_wmma_bf16<false,true,true,false>
            <<<dim3(DM / 128, MTOK / 128), blk, 0, stream>>>(
                HB, W2B + (size_t)i * DM * DFF, ffn_b2 + (size_t)i * DM, X, X, nullptr, DM, DFF);
    }
    rmsnorm_f32<<<MTOK, blk, 0, stream>>>(X, final_w, (float*)d_out, DM);
}
